// CWRRTEWindowCell_39659728011385
// MI455X (gfx1250) — compile-verified, matrix operand order checked by reading, not run on hardware
//
#include <hip/hip_runtime.h>
#include <hip/hip_bf16.h>

typedef __bf16 bf16_t;
typedef __attribute__((ext_vector_type(16))) __bf16 v16bf;
typedef __attribute__((ext_vector_type(8)))  __bf16 v8bf;
typedef __attribute__((ext_vector_type(8)))  float  v8f;
typedef __attribute__((ext_vector_type(4)))  unsigned int u32x4;
typedef __attribute__((ext_vector_type(8)))  int i32x8;
typedef __attribute__((ext_vector_type(4)))  int i32x4;

constexpr int cB = 4, cT = 1024, cO = 128, cD = 512, cH = 8, cHD = 64, cL = 6;
constexpr int cTK = cO + 2 * cT;          // 2176
constexpr int cD4 = 4 * cD, cD8 = 8 * cD; // 2048, 4096
constexpr int cBT = cB * cT;              // 4096
constexpr int cBTK = cB * cTK;            // 8704
constexpr int cBO = cB * cO;              // 512
constexpr unsigned cM = 100000u;

// ---------------------------------------------------------------- helpers
__device__ __forceinline__ v16bf load_frag(const bf16_t* p) {
  // A/B fragment, 16-bit, 16x32: halves j=0..7 -> K = base + j,
  // halves j=8..15 -> K = base + 16 + (j-8); base = k0 + (lane>=16 ? 8 : 0)
  v8bf lo = *(const v8bf*)p;
  v8bf hi = *(const v8bf*)(p + 16);
  v16bf r;
#pragma unroll
  for (int j = 0; j < 8; ++j) { r[j] = lo[j]; r[j + 8] = hi[j]; }
  return r;
}

__device__ __forceinline__ v8f wmma_bf16(v16bf a, v16bf b, v8f c) {
  return __builtin_amdgcn_wmma_f32_16x16x32_bf16(false, a, false, b, (short)0, c,
                                                 false, false);
}

__device__ __forceinline__ float wsum32(float v) {
#pragma unroll
  for (int m = 1; m < 32; m <<= 1) v += __shfl_xor(v, m, 32);
  return v;
}
__device__ __forceinline__ float xsum16(float v) {
#pragma unroll
  for (int m = 1; m < 16; m <<= 1) v += __shfl_xor(v, m, 32);
  return v;
}
__device__ __forceinline__ float xmax16(float v) {
#pragma unroll
  for (int m = 1; m < 16; m <<= 1) v = fmaxf(v, __shfl_xor(v, m, 32));
  return v;
}
__device__ __forceinline__ float sigf(float x) { return 1.0f / (1.0f + __expf(-x)); }

// TDM: DMA a (rows x 32) bf16 tile (row stride = K elements) from global into
// LDS (contiguous rows of 32 bf16). 2D tensor descriptor; groups 2/3 and the
// trailing group are unused (tile_dim2/3 == 0). 6-arg builtin (clang-23 form).
__device__ __forceinline__ void tdm_load_tile(const bf16_t* gptr, unsigned lds_off,
                                              int K, int dim0_left, int rows_left) {
  unsigned long long ga = (unsigned long long)(size_t)gptr;
  u32x4 g0;
  g0[0] = 1u;                                    // count=1, user descriptor
  g0[1] = lds_off;                               // lds_addr (bytes)
  g0[2] = (unsigned)(ga & 0xffffffffull);        // global_addr[31:0]
  g0[3] = (unsigned)((ga >> 32) & 0x01ffffffull) // global_addr[56:32]
          | 0x80000000u;                         // type=2 ("image")
  i32x8 g1;
  g1[0] = 0x00010000;                            // data_size=1 -> 2 bytes/elem
  g1[1] = (int)(((unsigned)dim0_left & 0xffffu) << 16);          // tensor_dim0 lo
  g1[2] = (int)((((unsigned)dim0_left >> 16) & 0xffffu)          // tensor_dim0 hi
                | (((unsigned)rows_left & 0xffffu) << 16));      // tensor_dim1 lo
  g1[3] = (int)((((unsigned)rows_left >> 16) & 0xffffu)          // tensor_dim1 hi
                | (32u << 16));                                  // tile_dim0 = 32
  g1[4] = 64;                                    // tile_dim1 = 64, tile_dim2 = 0
  g1[5] = K;                                     // tensor_dim0_stride[31:0]
  g1[6] = 0;                                     // stride hi / dim1_stride lo
  g1[7] = 0;
  i32x4 z4 = {0, 0, 0, 0};
  i32x8 z8 = {0, 0, 0, 0, 0, 0, 0, 0};
  __builtin_amdgcn_tensor_load_to_lds(g0, g1, z4, z4, z8, 0);
}

// ---------------------------------------------------------------- converts
__global__ void k_convert(const float* __restrict__ src, bf16_t* __restrict__ dst, int n) {
  int i = blockIdx.x * blockDim.x + threadIdx.x;
  if (i < n) dst[i] = (bf16_t)src[i];
}

// src (K,N) row-major f32 -> dst (N,K) row-major bf16 (W^T for B-fragment loads)
__global__ void k_convert_T(const float* __restrict__ src, bf16_t* __restrict__ dst,
                            int K, int N) {
  int i = blockIdx.x * blockDim.x + threadIdx.x;
  if (i >= K * N) return;
  int k = i / N, n = i - k * N;
  dst[(size_t)n * K + k] = (bf16_t)src[i];
}

// ---------------------------------------------------------------- prep kernels
__global__ void k_ssum_proj(const float* __restrict__ ssum, const float* __restrict__ W,
                            float* __restrict__ out) {
  int i = blockIdx.x * blockDim.x + threadIdx.x;
  if (i >= cB * cD) return;
  int b = i / cD, d = i - b * cD;
  float acc = 0.f;
  for (int k = 0; k < cD; ++k) acc += ssum[b * cD + k] * W[k * cD + d];
  out[i] = acc;
}

__global__ void k_build_x(const int* __restrict__ tokens, const float* __restrict__ embed,
                          const float* __restrict__ sproj, const float* __restrict__ alpha_gate,
                          float* __restrict__ cx, float* __restrict__ kv_f,
                          bf16_t* __restrict__ kv_b) {
  int i = blockIdx.x * blockDim.x + threadIdx.x;
  if (i >= cBT * cD) return;
  int d = i & (cD - 1);
  int bt = i / cD;
  int b = bt / cT, t = bt - b * cT;
  int tok = tokens[bt];
  float v = embed[(size_t)tok * cD + d] + sproj[b * cD + d] * sigf(alpha_gate[d]);
  cx[i] = v;
  size_t kvi = ((size_t)b * cTK + cO + cT + t) * cD + d;
  kv_f[kvi] = v;
  kv_b[kvi] = (bf16_t)v;
}

// one wave per (b,t): ngram hash -> gather -> gate -> rmsnorm -> kv rows [O, O+T)
__global__ __launch_bounds__(128) void k_engram(
    const int* __restrict__ tokens, const int* __restrict__ mem_ids,
    const float* __restrict__ table, const float* __restrict__ gate,
    const float* __restrict__ scale, float* __restrict__ kv_f, bf16_t* __restrict__ kv_b) {
  int lane = threadIdx.x & 31, wave = threadIdx.x >> 5;
  int row = blockIdx.x * 4 + wave;
  if (row >= cBT) return;
  int b = row / cT, t = row - b * cT;

  unsigned idxh[4];
#pragma unroll
  for (int h = 0; h < 4; ++h) {
    unsigned x = 131u + (unsigned)h * 1009u;
    unsigned pr[4];
#pragma unroll
    for (int i = 0; i < 4; ++i) { pr[i] = x; x = x * 31u + 1u; }
    unsigned hs = 0u;
#pragma unroll
    for (int i = 0; i < 4; ++i) {
      int p = t - i;
      unsigned tok = (p >= 0) ? (unsigned)tokens[b * cT + p]
                              : (unsigned)mem_ids[b * cO + cO + p];
      hs += tok * pr[i];
    }
    idxh[h] = hs % cM;
  }
  float v[16];
  float ss = 0.f;
#pragma unroll
  for (int i = 0; i < 16; ++i) {
    int d = lane + i * 32;
    int h = d >> 7, dd = d & 127;
    float g = sigf(gate[h * 128 + dd]);
    float val = table[((size_t)idxh[h] * 4 + h) * 128 + dd] * g;
    v[i] = val;
    ss += val * val;
  }
  ss = wsum32(ss);
  float inv = rsqrtf(ss / (float)cD + 1e-6f);
  size_t base = ((size_t)b * cTK + cO + t) * cD;
#pragma unroll
  for (int i = 0; i < 16; ++i) {
    int d = lane + i * 32;
    float y = v[i] * inv * scale[d];
    kv_f[base + d] = y;
    kv_b[base + d] = (bf16_t)y;
  }
}

// one wave per row; out_row = (row/irpb)*orpb + row%irpb (kv row remapping for mem_p)
__global__ __launch_bounds__(128) void k_rmsnorm(
    const float* __restrict__ in, const float* __restrict__ scale,
    float* __restrict__ of, bf16_t* __restrict__ ob, int nrows, int irpb, int orpb) {
  int lane = threadIdx.x & 31, wave = threadIdx.x >> 5;
  int row = blockIdx.x * 4 + wave;
  if (row >= nrows) return;
  const float* x = in + (size_t)row * cD;
  float v[16];
  float ss = 0.f;
#pragma unroll
  for (int i = 0; i < 16; ++i) { v[i] = x[lane + i * 32]; ss += v[i] * v[i]; }
  ss = wsum32(ss);
  float inv = rsqrtf(ss / (float)cD + 1e-6f);
  size_t orow = (size_t)(row / irpb) * orpb + (row % irpb);
#pragma unroll
  for (int i = 0; i < 16; ++i) {
    int d = lane + i * 32;
    float y = v[i] * inv * scale[d];
    if (of) of[orow * cD + d] = y;
    if (ob) ob[orow * cD + d] = (bf16_t)y;
  }
}

// ---------------------------------------------------------------- WMMA GEMM
// C(M,N) = A(M,K)bf16 @ Wt(N,K)bf16^T + bias; per-wave 16x64 tile.
// All 4 waves of a block share n0, so the 64x32 B tile is staged into LDS by
// the Tensor Data Mover (double-buffered), synced with s_wait_tensorcnt.
// Grids are launched so M % 64 == 0 (no partial blocks: barriers are uniform).
__global__ __launch_bounds__(128) void k_gemm(
    const bf16_t* __restrict__ A, const bf16_t* __restrict__ Wt,
    const float* __restrict__ bias, float* __restrict__ Cf, bf16_t* __restrict__ Cb,
    int M, int N, int K, int accumulate) {
  __shared__ bf16_t Bs[2][64][32]; // two 4KB B tiles
  int lane = threadIdx.x & 31, wave = threadIdx.x >> 5;
  int m0 = (blockIdx.y * 4 + wave) * 16;
  int n0 = blockIdx.x * 64;
  int half = lane >> 4, l16 = lane & 15;

  v8f acc[4];
#pragma unroll
  for (int f = 0; f < 4; ++f)
#pragma unroll
    for (int e = 0; e < 8; ++e) acc[f][e] = 0.f;

  const bf16_t* ap = A + (size_t)(m0 + l16) * K + half * 8;
  const bf16_t* btile = Wt + (size_t)n0 * K; // row n0, col 0
  unsigned lds0 = (unsigned)(size_t)&Bs[0][0][0];
  unsigned lds1 = (unsigned)(size_t)&Bs[1][0][0];

  if (wave == 0) { // prologue: DMA tiles 0 and 1
    tdm_load_tile(btile, lds0, K, K, N - n0);
    tdm_load_tile(btile + 32, lds1, K, K - 32, N - n0);
  }

  v16bf a = load_frag(ap); // A tile 0 (register double buffer)
  for (int k0 = 0; k0 < K; k0 += 32) {
    int buf = (k0 >> 5) & 1;
    v16bf an = a;
    if (k0 + 32 < K) an = load_frag(ap + k0 + 32); // next A in flight
    __builtin_prefetch(ap + k0 + 64, 0, 1);
    if (wave == 0) { // tile k0 done when at most the newer DMA is outstanding
      if (k0 + 32 < K) __builtin_amdgcn_s_wait_tensorcnt(1);
      else             __builtin_amdgcn_s_wait_tensorcnt(0);
    }
    __syncthreads(); // B tile `buf` visible to all waves
#pragma unroll
    for (int f = 0; f < 4; ++f) {
      v16bf bfr = load_frag(&Bs[buf][f * 16 + l16][half * 8]);
      acc[f] = wmma_bf16(a, bfr, acc[f]);
    }
    __syncthreads(); // all waves done reading `buf`
    if (wave == 0 && k0 + 64 < K)
      tdm_load_tile(btile + k0 + 64, buf == 0 ? lds0 : lds1, K, K - (k0 + 64),
                    N - n0);
    a = an;
  }
#pragma unroll
  for (int f = 0; f < 4; ++f) {
    int n = n0 + f * 16 + l16;
    float bv = bias ? bias[n] : 0.f;
#pragma unroll
    for (int r = 0; r < 8; ++r) {
      int m = m0 + r + half * 8;
      size_t idx = (size_t)m * N + n;
      float v = acc[f][r] + bv;
      if (accumulate) v += Cf[idx];
      if (Cf) Cf[idx] = v;
      if (Cb) Cb[idx] = (bf16_t)v;
    }
  }
}

// fused MLP hidden: act = silu(A@Win[:,n]) * (A@Win[:,n+4D]); Wt is (8D,K)
__global__ __launch_bounds__(128) void k_mlp(
    const bf16_t* __restrict__ A, const bf16_t* __restrict__ Wt,
    bf16_t* __restrict__ act, int M, int K) {
  int lane = threadIdx.x & 31, wave = threadIdx.x >> 5;
  int m0 = (blockIdx.y * 4 + wave) * 16;
  int n0 = blockIdx.x * 64;
  if (m0 >= M) return;
  int half = lane >> 4, l16 = lane & 15;

  v8f ag[4], au[4];
#pragma unroll
  for (int f = 0; f < 4; ++f)
#pragma unroll
    for (int e = 0; e < 8; ++e) { ag[f][e] = 0.f; au[f][e] = 0.f; }

  const bf16_t* ap = A + (size_t)(m0 + l16) * K + half * 8;
  const bf16_t* gp[4];
  const bf16_t* up[4];
#pragma unroll
  for (int f = 0; f < 4; ++f) {
    gp[f] = Wt + (size_t)(n0 + f * 16 + l16) * K + half * 8;
    up[f] = Wt + (size_t)(cD4 + n0 + f * 16 + l16) * K + half * 8;
  }
  v16bf a = load_frag(ap);
  for (int k0 = 0; k0 < K; k0 += 32) {
    v16bf an = a;
    if (k0 + 32 < K) an = load_frag(ap + k0 + 32);
#pragma unroll
    for (int f = 0; f < 4; ++f) {
      v16bf g = load_frag(gp[f] + k0);
      ag[f] = wmma_bf16(a, g, ag[f]);
      v16bf u = load_frag(up[f] + k0);
      au[f] = wmma_bf16(a, u, au[f]);
    }
    a = an;
  }
#pragma unroll
  for (int f = 0; f < 4; ++f) {
    int n = n0 + f * 16 + l16;
#pragma unroll
    for (int r = 0; r < 8; ++r) {
      int m = m0 + r + half * 8;
      float g = ag[f][r], u = au[f][r];
      act[(size_t)m * cD4 + n] = (bf16_t)(g * sigf(g) * u);
    }
  }
}

// ---------------------------------------------------------------- RoPE + V transpose
// in/out layout (rows, H, HD); pos = row % pos_mod
__global__ void k_rope(const float* __restrict__ in, bf16_t* __restrict__ out,
                       int rows, int pos_mod) {
  int i = blockIdx.x * blockDim.x + threadIdx.x;
  if (i >= rows * cH * 32) return;
  int pair = i & 31;
  int rh = i >> 5;
  int h = rh & 7, row = rh >> 3;
  int pos = row % pos_mod;
  float fr = __powf(10000.0f, -(float)(2 * pair) / (float)cHD);
  float s, c;
  __sincosf((float)pos * fr, &s, &c);
  size_t base = ((size_t)row * cH + h) * cHD + 2 * pair;
  float x0 = in[base], x1 = in[base + 1];
  out[base] = (bf16_t)(x0 * c - x1 * s);
  out[base + 1] = (bf16_t)(x0 * s + x1 * c);
}

// v_f32 (B,TK,H,HD) -> vT bf16 (B,H,HD,TK) so P@V B-fragments load contiguously
__global__ void k_transpose_v(const float* __restrict__ in, bf16_t* __restrict__ out) {
  int i = blockIdx.x * blockDim.x + threadIdx.x;
  if (i >= cBTK * cD) return;
  int d = i & (cD - 1);
  int bt = i / cD;
  int b = bt / cTK, t = bt - b * cTK;
  int h = d >> 6, hd = d & 63;
  out[(((size_t)b * cH + h) * cHD + hd) * cTK + t] = (bf16_t)in[i];
}

// ---------------------------------------------------------------- flash attention
// one wave = 16 query rows x full TK; WMMA for scores and P@V; online softmax.
__global__ __launch_bounds__(128) void k_attn(
    const bf16_t* __restrict__ Q,   // (B,T,H,HD) roped
    const bf16_t* __restrict__ Km,  // (B,TK,H,HD) roped
    const bf16_t* __restrict__ Vt,  // (B,H,HD,TK)
    const int* __restrict__ tokens, // (B,T)
    bf16_t* __restrict__ Out) {     // (B,T,D)
  __shared__ bf16_t plds[4][16][32];
  int lane = threadIdx.x & 31, wave = threadIdx.x >> 5;
  int bh = blockIdx.y;
  int b = bh >> 3, h = bh & 7;
  int t0 = (blockIdx.x * 4 + wave) * 16;
  int half = lane >> 4, l16 = lane & 15;
  const float scale = 0.125f; // 1/sqrt(64)

  const bf16_t* qrow = Q + (((size_t)b * cT + t0 + l16) * cH + h) * cHD + half * 8;
  v16bf qf0 = load_frag(qrow);
  v16bf qf1 = load_frag(qrow + 32);

  v8f acc[4];
#pragma unroll
  for (int f = 0; f < 4; ++f)
#pragma unroll
    for (int e = 0; e < 8; ++e) acc[f][e] = 0.f;
  float rm[8], rl[8];
#pragma unroll
  for (int r = 0; r < 8; ++r) { rm[r] = -1e30f; rl[r] = 0.f; }

  for (int kb0 = 0; kb0 < cTK; kb0 += 32) {
    float sc[2][8];
#pragma unroll
    for (int kc = 0; kc < 2; ++kc) {
      int key = kb0 + kc * 16 + l16;
      const bf16_t* krow = Km + (((size_t)b * cTK + key) * cH + h) * cHD + half * 8;
      v16bf b0 = load_frag(krow);
      v16bf b1 = load_frag(krow + 32);
      v8f sf;
#pragma unroll
      for (int e = 0; e < 8; ++e) sf[e] = 0.f;
      sf = wmma_bf16(qf0, b0, sf);
      sf = wmma_bf16(qf1, b1, sf);
#pragma unroll
      for (int r = 0; r < 8; ++r) {
        int row_t = t0 + r + half * 8;
        bool ok;
        if (key < cO) {
          ok = true;
        } else {
          int j = key - cO;
          if (j >= cT) j -= cT;
          ok = (j <= row_t) && (tokens[b * cT + j] != 0);
        }
        sc[kc][r] = ok ? sf[r] * scale : -1e9f;
      }
    }
    float cmax[8], csum[8];
#pragma unroll
    for (int r = 0; r < 8; ++r) cmax[r] = xmax16(fmaxf(sc[0][r], sc[1][r]));
#pragma unroll
    for (int r = 0; r < 8; ++r) {
      float nm = fmaxf(rm[r], cmax[r]);
      float corr = __expf(rm[r] - nm);
      rm[r] = nm;
      rl[r] *= corr;
#pragma unroll
      for (int f = 0; f < 4; ++f) acc[f][r] *= corr;
      csum[r] = 0.f;
    }
#pragma unroll
    for (int kc = 0; kc < 2; ++kc)
#pragma unroll
      for (int r = 0; r < 8; ++r) {
        float p = __expf(sc[kc][r] - rm[r]);
        csum[r] += p;
        plds[wave][r + half * 8][kc * 16 + l16] = (bf16_t)p;
      }
#pragma unroll
    for (int r = 0; r < 8; ++r) rl[r] += xsum16(csum[r]);

    // re-stripe P (C layout) -> A fragment via LDS
    v16bf pa = load_frag(&plds[wave][l16][half * 8]);
#pragma unroll
    for (int f = 0; f < 4; ++f) {
      const bf16_t* vrow =
          Vt + (((size_t)b * cH + h) * cHD + f * 16 + l16) * cTK + kb0 + half * 8;
      v16bf vb = load_frag(vrow);
      acc[f] = wmma_bf16(pa, vb, acc[f]);
    }
  }
#pragma unroll
  for (int f = 0; f < 4; ++f)
#pragma unroll
    for (int r = 0; r < 8; ++r) {
      size_t o = ((size_t)b * cT + t0 + r + half * 8) * cD + h * cHD + f * 16 + l16;
      Out[o] = (bf16_t)(acc[f][r] / rl[r]);
    }
}

// ---------------------------------------------------------------- host orchestration
extern "C" void kernel_launch(void* const* d_in, const int* in_sizes, int n_in,
                              void* d_out, int out_size, void* d_ws, size_t ws_size,
                              hipStream_t stream) {
  (void)in_sizes; (void)n_in; (void)out_size; (void)ws_size;
  const int* tokens = (const int*)d_in[0];
  const int* mem_ids = (const int*)d_in[1];
  const float* mem_emb = (const float*)d_in[2];
  const float* ssum = (const float*)d_in[3];
  const float* embed = (const float*)d_in[4];
  const float* engram_table = (const float*)d_in[5];
  const float* engram_gate = (const float*)d_in[6];
  const float* engram_scale = (const float*)d_in[7];
  const float* mem_W = (const float*)d_in[8];
  const float* mem_b = (const float*)d_in[9];
  const float* mem_scale = (const float*)d_in[10];
  const float* alpha_gate = (const float*)d_in[11];
  const float* ssum_W = (const float*)d_in[12];
  const float* ln1 = (const float*)d_in[13];
  const float* qW = (const float*)d_in[14];
  const float* qb = (const float*)d_in[15];
  const float* kW = (const float*)d_in[16];
  const float* kb = (const float*)d_in[17];
  const float* vW = (const float*)d_in[18];
  const float* vb = (const float*)d_in[19];
  const float* oW = (const float*)d_in[20];
  const float* ob = (const float*)d_in[21];
  const float* ln2 = (const float*)d_in[22];
  const float* w_in = (const float*)d_in[23];
  const float* w_out = (const float*)d_in[24];
  const float* final_scale = (const float*)d_in[25];
  float* outp = (float*)d_out;

  char* ws = (char*)d_ws;
  size_t off = 0;
  auto alloc = [&](size_t bytes) -> void* {
    void* p = ws + off;
    off += (bytes + 255) & ~(size_t)255;
    return p;
  };

  // bf16 transposed weights (~52MB)
  bf16_t* qWt = (bf16_t*)alloc((size_t)cL * cD * cD * 2);
  bf16_t* kWt = (bf16_t*)alloc((size_t)cL * cD * cD * 2);
  bf16_t* vWt = (bf16_t*)alloc((size_t)cL * cD * cD * 2);
  bf16_t* oWt = (bf16_t*)alloc((size_t)cL * cD * cD * 2);
  bf16_t* wiT = (bf16_t*)alloc((size_t)cL * cD * cD8 * 2);
  bf16_t* woT = (bf16_t*)alloc((size_t)cL * cD4 * cD * 2);
  bf16_t* mWt = (bf16_t*)alloc((size_t)cD * cD * 2);
  bf16_t* memb_b = (bf16_t*)alloc((size_t)cBO * cD * 2);
  // activations (~110MB; v_f32 region reused for the MLP act buffer)
  float* kv_f = (float*)alloc((size_t)cBTK * cD * 4);
  bf16_t* kv_b = (bf16_t*)alloc((size_t)cBTK * cD * 2);
  float* cx = (float*)alloc((size_t)cBT * cD * 4);
  bf16_t* nx_b = (bf16_t*)alloc((size_t)cBT * cD * 2);
  float* sproj = (float*)alloc((size_t)cB * cD * 4);
  float* tmp_mp = (float*)alloc((size_t)cBO * cD * 4);
  float* q_f = (float*)alloc((size_t)cBT * cD * 4);
  float* k_f = (float*)alloc((size_t)cBTK * cD * 4);
  float* v_f = (float*)alloc((size_t)cBTK * cD * 4);
  bf16_t* q_b = (bf16_t*)alloc((size_t)cBT * cD * 2);
  bf16_t* k_b = (bf16_t*)alloc((size_t)cBTK * cD * 2);
  bf16_t* vT_b = (bf16_t*)alloc((size_t)cBTK * cD * 2);
  bf16_t* at_b = (bf16_t*)alloc((size_t)cBT * cD * 2);
  bf16_t* act_b = (bf16_t*)v_f; // alias: v_f32 dead after transpose, act written later

  auto cvtT = [&](const float* s, bf16_t* d, int K, int N) {
    int n = K * N;
    k_convert_T<<<(n + 255) / 256, 256, 0, stream>>>(s, d, K, N);
  };
  for (int l = 0; l < cL; ++l) {
    size_t dd = (size_t)l * cD * cD;
    cvtT(qW + dd, qWt + dd, cD, cD);
    cvtT(kW + dd, kWt + dd, cD, cD);
    cvtT(vW + dd, vWt + dd, cD, cD);
    cvtT(oW + dd, oWt + dd, cD, cD);
    cvtT(w_in + (size_t)l * cD * cD8, wiT + (size_t)l * cD * cD8, cD, cD8);
    cvtT(w_out + (size_t)l * cD4 * cD, woT + (size_t)l * cD4 * cD, cD4, cD);
  }
  cvtT(mem_W, mWt, cD, cD);
  k_convert<<<(cBO * cD + 255) / 256, 256, 0, stream>>>(mem_emb, memb_b, cBO * cD);

  // prologue: x / engram / mem_p -> kv_seq
  k_ssum_proj<<<(cB * cD + 255) / 256, 256, 0, stream>>>(ssum, ssum_W, sproj);
  k_build_x<<<(cBT * cD + 255) / 256, 256, 0, stream>>>(tokens, embed, sproj,
                                                        alpha_gate, cx, kv_f, kv_b);
  k_engram<<<cBT / 4, 128, 0, stream>>>(tokens, mem_ids, engram_table, engram_gate,
                                        engram_scale, kv_f, kv_b);
  k_gemm<<<dim3(cD / 64, cBO / 64), 128, 0, stream>>>(memb_b, mWt, mem_b, tmp_mp,
                                                      nullptr, cBO, cD, cD, 0);
  k_rmsnorm<<<cBO / 4, 128, 0, stream>>>(tmp_mp, mem_scale, kv_f, kv_b, cBO, cO, cTK);

  // layers
  for (int l = 0; l < cL; ++l) {
    size_t dd = (size_t)l * cD * cD;
    k_rmsnorm<<<cBT / 4, 128, 0, stream>>>(cx, ln1 + l * cD, nullptr, nx_b, cBT, cT, cT);
    k_gemm<<<dim3(cD / 64, cBT / 64), 128, 0, stream>>>(nx_b, qWt + dd, qb + l * cD,
                                                        q_f, nullptr, cBT, cD, cD, 0);
    k_gemm<<<dim3(cD / 64, cBTK / 64), 128, 0, stream>>>(kv_b, kWt + dd, kb + l * cD,
                                                         k_f, nullptr, cBTK, cD, cD, 0);
    k_gemm<<<dim3(cD / 64, cBTK / 64), 128, 0, stream>>>(kv_b, vWt + dd, vb + l * cD,
                                                         v_f, nullptr, cBTK, cD, cD, 0);
    k_rope<<<(cBT * cH * 32 + 255) / 256, 256, 0, stream>>>(q_f, q_b, cBT, cT);
    k_rope<<<(cBTK * cH * 32 + 255) / 256, 256, 0, stream>>>(k_f, k_b, cBTK, cTK);
    k_transpose_v<<<(cBTK * cD + 255) / 256, 256, 0, stream>>>(v_f, vT_b);
    k_attn<<<dim3(cT / 64, cB * cH), 128, 0, stream>>>(q_b, k_b, vT_b, tokens, at_b);
    k_gemm<<<dim3(cD / 64, cBT / 64), 128, 0, stream>>>(at_b, oWt + dd, ob + l * cD,
                                                        cx, nullptr, cBT, cD, cD, 1);
    k_rmsnorm<<<cBT / 4, 128, 0, stream>>>(cx, ln2 + l * cD, nullptr, nx_b, cBT, cT, cT);
    k_mlp<<<dim3(cD4 / 64, cBT / 64), 128, 0, stream>>>(nx_b, wiT + (size_t)l * cD * cD8,
                                                        act_b, cBT, cD);
    k_gemm<<<dim3(cD / 64, cBT / 64), 128, 0, stream>>>(act_b, woT + (size_t)l * cD4 * cD,
                                                        nullptr, cx, nullptr, cBT, cD,
                                                        cD4, 1);
  }
  k_rmsnorm<<<cBT / 4, 128, 0, stream>>>(cx, final_scale, outp, nullptr, cBT, cT, cT);
}
